// MultiHeadAttention_78099685310911
// MI455X (gfx1250) — matrix operand order repaired and checked by reading, not compiled, not run on hardware
//
#include <hip/hip_runtime.h>

#define S_LEN  4096
#define DMODEL 1024
#define NHEADS 16
#define DK     64

typedef __attribute__((ext_vector_type(16))) __bf16 v16bf;
typedef __attribute__((ext_vector_type(8)))  __bf16 v8bf;
typedef __attribute__((ext_vector_type(8)))  float  v8f;
typedef __attribute__((ext_vector_type(4)))  int    v4i;

union AFrag { v16bf v; v8bf h[2]; };

// ---- feature probes --------------------------------------------------------
#if defined(__has_builtin)
#  if __has_builtin(__builtin_amdgcn_global_load_async_to_lds_b128)
#    define HAVE_ASYNC_LDS 1
#  else
#    define HAVE_ASYNC_LDS 0
#  endif
#  if __has_builtin(__builtin_amdgcn_s_wait_asynccnt)
#    define HAVE_WAIT_ASYNC_BUILTIN 1
#  else
#    define HAVE_WAIT_ASYNC_BUILTIN 0
#  endif
#else
#  define HAVE_ASYNC_LDS 0
#  define HAVE_WAIT_ASYNC_BUILTIN 0
#endif

__device__ __forceinline__ unsigned short f2bf(float f) {
    unsigned int u = __float_as_uint(f);
    u += 0x7FFFu + ((u >> 16) & 1u);      // round-to-nearest-even
    return (unsigned short)(u >> 16);
}

// DPP16 row-rotate butterfly reductions over the 16-lane row (wave32: two
// DPP rows per wave, matching the 16-lane N-groups of the WMMA C layout).
#define DPP_ROR(x, n)                                                          \
    __int_as_float(__builtin_amdgcn_update_dpp(                                \
        0, __float_as_int(x), 0x120 + (n), 0xF, 0xF, true))

__device__ __forceinline__ float rowmax16(float x) {
    float t;
    t = DPP_ROR(x, 1); x = fmaxf(x, t);
    t = DPP_ROR(x, 2); x = fmaxf(x, t);
    t = DPP_ROR(x, 4); x = fmaxf(x, t);
    t = DPP_ROR(x, 8); x = fmaxf(x, t);
    return x;
}
__device__ __forceinline__ float rowsum16(float x) {
    float t;
    t = DPP_ROR(x, 1); x += t;
    t = DPP_ROR(x, 2); x += t;
    t = DPP_ROR(x, 4); x += t;
    t = DPP_ROR(x, 8); x += t;
    return x;
}

// 16-byte global -> LDS copy: async (ASYNCcnt-tracked) when available.
__device__ __forceinline__ void copy16(unsigned short* l, const unsigned short* g) {
#if HAVE_ASYNC_LDS
    __builtin_amdgcn_global_load_async_to_lds_b128(
        (__attribute__((address_space(1))) v4i*)(unsigned short*)g,
        (__attribute__((address_space(3))) v4i*)l, 0, 0);
#else
    *(int4*)l = *(const int4*)g;
#endif
}

template <int N>
__device__ __forceinline__ void wait_async() {
#if HAVE_ASYNC_LDS
#  if HAVE_WAIT_ASYNC_BUILTIN
    __builtin_amdgcn_s_wait_asynccnt((unsigned short)N);
#  else
    asm volatile("s_wait_asynccnt %0" ::"i"(N));
#  endif
#endif
}

// ---------------------------------------------------------------------------
// fp32 -> bf16 conversion (one-shot)
// ---------------------------------------------------------------------------
__global__ __launch_bounds__(256) void cvt_f32_bf16(
    const float* __restrict__ in, unsigned short* __restrict__ out, int n) {
    int i = (blockIdx.x * blockDim.x + threadIdx.x) * 4;
    if (i + 3 < n) {
        float4 f = *(const float4*)(in + i);
        out[i + 0] = f2bf(f.x);
        out[i + 1] = f2bf(f.y);
        out[i + 2] = f2bf(f.z);
        out[i + 3] = f2bf(f.w);
    }
}

// ---------------------------------------------------------------------------
// GEMM: C[M,N] = A[M,K] * B[N,K]^T, bf16 WMMA, fp32 accum.
// Wave tile = 32(M) x 64(N): 2 A-fragments x 4 B-fragments = 8 independent
// WMMAs per 32-K step (2x B reuse, deep latency hiding). Block = 8 waves ->
// 256(M) x 64(N) per block.
// MODE: 0 = bf16 row-major out, 1 = bf16 transposed out ([N,M]), 2 = f32 out.
// ---------------------------------------------------------------------------
template <int MODE>
__global__ __launch_bounds__(256, 1) void gemm_bf16_xwT(
    const unsigned short* __restrict__ A, const unsigned short* __restrict__ B,
    void* __restrict__ C, int M, int N, int K) {
    const int lane = threadIdx.x & 31;
    const int wave = threadIdx.x >> 5;
    const int half = lane >> 4;
    const int l16  = lane & 15;
    const int m0 = blockIdx.y * 256 + wave * 32;
    const int n0 = blockIdx.x * 64;

    v8f acc[8] = {};
    const unsigned short* arow0 = A + (size_t)(m0 + l16) * K;
    const unsigned short* arow1 = A + (size_t)(m0 + 16 + l16) * K;

    for (int k0 = 0; k0 < K; k0 += 32) {
        AFrag a0, a1;
        a0.h[0] = *(const v8bf*)(arow0 + k0 + half * 16);
        a0.h[1] = *(const v8bf*)(arow0 + k0 + half * 16 + 8);
        a1.h[0] = *(const v8bf*)(arow1 + k0 + half * 16);
        a1.h[1] = *(const v8bf*)(arow1 + k0 + half * 16 + 8);
#pragma unroll
        for (int nt = 0; nt < 4; ++nt) {
            const unsigned short* bp =
                B + (size_t)(n0 + nt * 16 + l16) * K + k0 + half * 16;
            v16bf b = *(const v16bf*)bp;
            acc[nt] = __builtin_amdgcn_wmma_f32_16x16x32_bf16(
                false, a0.v, false, b, (short)0, acc[nt], false, false);
            acc[4 + nt] = __builtin_amdgcn_wmma_f32_16x16x32_bf16(
                false, a1.v, false, b, (short)0, acc[4 + nt], false, false);
        }
    }

#pragma unroll
    for (int mt = 0; mt < 2; ++mt)
#pragma unroll
        for (int nt = 0; nt < 4; ++nt)
#pragma unroll
            for (int r = 0; r < 8; ++r) {
                const int m = m0 + mt * 16 + half * 8 + r;
                const int n = n0 + nt * 16 + l16;
                const float val = acc[mt * 4 + nt][r];
                if (MODE == 2)
                    ((float*)C)[(size_t)m * N + n] = val;
                else if (MODE == 0)
                    ((unsigned short*)C)[(size_t)m * N + n] = f2bf(val);
                else
                    ((unsigned short*)C)[(size_t)n * M + m] = f2bf(val);
            }
}

// ---------------------------------------------------------------------------
// Flash attention (causal), bf16 WMMA, fp32 accum, online softmax.
// Grid: (S/128 query blocks, NHEADS). Block: 256 threads = 8 waves, each wave
// owns 16 query rows (128 rows/block). All 8 waves consume the SAME K/V tile
// per iteration: tiles staged once per block into LDS via async load-to-LDS
// (ASYNCcnt), double-buffered so key block kb+1 streams in during kb's
// compute. Rows padded to 72 elems (144 B) for 16 B alignment + bank spread.
// ---------------------------------------------------------------------------
__device__ __forceinline__ void stage_tiles(
    unsigned short (&kt)[64][72], unsigned short (&vt)[64][72],
    const unsigned short* __restrict__ Kb, const unsigned short* __restrict__ Vt,
    int key0, int h, int tid) {
#pragma unroll
    for (int i = 0; i < 2; ++i) {
        const int c = tid + i * 256;        // 512 16B-chunks per tile
        const int r = c >> 3, part = c & 7;
        copy16(&kt[r][part * 8],
               Kb + (size_t)(key0 + r) * DMODEL + h * DK + part * 8);
        copy16(&vt[r][part * 8],
               Vt + (size_t)(h * DK + r) * S_LEN + key0 + part * 8);
    }
}

__global__ __launch_bounds__(256, 1) void flash_attn_kernel(
    const unsigned short* __restrict__ Qb, const unsigned short* __restrict__ Kb,
    const unsigned short* __restrict__ Vt, unsigned short* __restrict__ Cb) {
    const int tid  = threadIdx.x;
    const int lane = tid & 31;
    const int wid  = tid >> 5;            // 0..7
    const int half = lane >> 4;
    const int l16  = lane & 15;
    const int h    = blockIdx.y;
    const int qb   = blockIdx.x;          // 128-row query block
    const int q0   = qb * 128 + wid * 16; // this wave's first query row
    const int kbmax = 2 * qb + 1;         // causal: keys <= last query row

    __shared__ __align__(16) unsigned short ktile[2][64][72];  // 2 x 9 KB
    __shared__ __align__(16) unsigned short vtile[2][64][72];  // 2 x 9 KB
    __shared__ __align__(16) unsigned short pshare[8][16][72]; // 18 KB

    // Q A-fragments, loaded once (16 rows x 64 dk = two 16x32 fragments)
    AFrag qf[2];
    {
        const unsigned short* qrow = Qb + (size_t)(q0 + l16) * DMODEL + h * DK;
#pragma unroll
        for (int kk = 0; kk < 2; ++kk) {
            qf[kk].h[0] = *(const v8bf*)(qrow + kk * 32 + half * 16);
            qf[kk].h[1] = *(const v8bf*)(qrow + kk * 32 + half * 16 + 8);
        }
    }

    v8f o[4] = {};
    float mrow[8], lrow[8];
#pragma unroll
    for (int r = 0; r < 8; ++r) { mrow[r] = -1e30f; lrow[r] = 0.0f; }

    stage_tiles(ktile[0], vtile[0], Kb, Vt, 0, h, tid);   // prologue: kb = 0

    for (int kb = 0; kb <= kbmax; ++kb) {
        const int buf  = kb & 1;
        const int key0 = kb * 64;

        if (kb < kbmax) {   // prefetch next key block into the other buffer
            stage_tiles(ktile[buf ^ 1], vtile[buf ^ 1], Kb, Vt,
                        (kb + 1) * 64, h, tid);
            wait_async<4>();   // current buffer's 4 async ops retired
        } else {
            wait_async<0>();
        }
        __syncthreads();

        // ---- S = Q * K^T : 8 WMMAs, B fragments from LDS ----------------
        v8f s[4] = {};
#pragma unroll
        for (int kk = 0; kk < 2; ++kk) {
#pragma unroll
            for (int nt = 0; nt < 4; ++nt) {
                v16bf b = *(const v16bf*)&ktile[buf][nt * 16 + l16]
                                               [kk * 32 + half * 16];
                s[nt] = __builtin_amdgcn_wmma_f32_16x16x32_bf16(
                    false, qf[kk].v, false, b, (short)0, s[nt], false, false);
            }
        }

        // ---- scale, causal mask, online softmax (DPP row reductions) ----
        const int mykey = key0 + l16;
#pragma unroll
        for (int r = 0; r < 8; ++r) {
            const int qrow_g = q0 + half * 8 + r;
            float mx = -1e30f;
#pragma unroll
            for (int nt = 0; nt < 4; ++nt) {
                float e = s[nt][r] * 0.125f;               // 1/sqrt(64)
                if (mykey + nt * 16 > qrow_g) e = -1.0e9f; // causal
                s[nt][r] = e;
                mx = fmaxf(mx, e);
            }
            mx = rowmax16(mx);
            const float mnew  = fmaxf(mrow[r], mx);
            const float alpha = __expf(mrow[r] - mnew);
            mrow[r] = mnew;
            float rs = 0.0f;
#pragma unroll
            for (int nt = 0; nt < 4; ++nt) {
                const float p = __expf(s[nt][r] - mnew);
                rs += p;
                pshare[wid][half * 8 + r][nt * 16 + l16] = f2bf(p);
            }
            rs = rowsum16(rs);
            lrow[r] = lrow[r] * alpha + rs;
#pragma unroll
            for (int nt = 0; nt < 4; ++nt) o[nt][r] *= alpha;
        }

        // wave-private LDS RAW: drain P stores before transposed reload
        asm volatile("s_wait_dscnt 0" ::: "memory");

        // ---- O += P * V : 8 WMMAs, V fragments from LDS -----------------
#pragma unroll
        for (int kk = 0; kk < 2; ++kk) {
            AFrag p;
            const unsigned short* prow = &pshare[wid][l16][kk * 32];
            p.h[0] = *(const v8bf*)(prow + half * 16);
            p.h[1] = *(const v8bf*)(prow + half * 16 + 8);
#pragma unroll
            for (int nt = 0; nt < 4; ++nt) {
                v16bf b = *(const v16bf*)&vtile[buf][nt * 16 + l16]
                                               [kk * 32 + half * 16];
                o[nt] = __builtin_amdgcn_wmma_f32_16x16x32_bf16(
                    false, p.v, false, b, (short)0, o[nt], false, false);
            }
        }
        __syncthreads();   // all waves done with buf before it is restaged
    }

    // ---- normalize & write context (bf16) -------------------------------
#pragma unroll
    for (int r = 0; r < 8; ++r) {
        const float inv = 1.0f / lrow[r];
        const int row = q0 + half * 8 + r;
#pragma unroll
        for (int nt = 0; nt < 4; ++nt)
            Cb[(size_t)row * DMODEL + h * DK + nt * 16 + l16] =
                f2bf(o[nt][r] * inv);
    }
}

// ---------------------------------------------------------------------------
extern "C" void kernel_launch(void* const* d_in, const int* in_sizes, int n_in,
                              void* d_out, int out_size, void* d_ws,
                              size_t ws_size, hipStream_t stream) {
    (void)in_sizes; (void)n_in; (void)out_size; (void)ws_size;

    const float* x  = (const float*)d_in[0];
    const float* Wq = (const float*)d_in[1];
    const float* Wk = (const float*)d_in[2];
    const float* Wv = (const float*)d_in[3];
    const float* Wo = (const float*)d_in[4];

    char* ws = (char*)d_ws;
    const size_t MB = 1024u * 1024u;
    unsigned short* xb  = (unsigned short*)(ws + 0 * MB);   // 8 MB  x bf16
    unsigned short* wqb = (unsigned short*)(ws + 8 * MB);   // 2 MB
    unsigned short* wkb = (unsigned short*)(ws + 10 * MB);  // 2 MB
    unsigned short* wvb = (unsigned short*)(ws + 12 * MB);  // 2 MB
    unsigned short* wob = (unsigned short*)(ws + 14 * MB);  // 2 MB
    unsigned short* Qb  = (unsigned short*)(ws + 16 * MB);  // 8 MB  [S][D]
    unsigned short* Kb  = (unsigned short*)(ws + 24 * MB);  // 8 MB  [S][D]
    unsigned short* Vt  = (unsigned short*)(ws + 32 * MB);  // 8 MB  [D][S]
    unsigned short* Cb  = (unsigned short*)(ws + 40 * MB);  // 8 MB  [S][D]

    const int nx = S_LEN * DMODEL;   // 4,194,304
    const int nw = DMODEL * DMODEL;  // 1,048,576
    cvt_f32_bf16<<<nx / 1024, 256, 0, stream>>>(x,  xb,  nx);
    cvt_f32_bf16<<<nw / 1024, 256, 0, stream>>>(Wq, wqb, nw);
    cvt_f32_bf16<<<nw / 1024, 256, 0, stream>>>(Wk, wkb, nw);
    cvt_f32_bf16<<<nw / 1024, 256, 0, stream>>>(Wv, wvb, nw);
    cvt_f32_bf16<<<nw / 1024, 256, 0, stream>>>(Wo, wob, nw);

    dim3 gg(DMODEL / 64, S_LEN / 256);  // (16, 16) blocks of 8 waves
    gemm_bf16_xwT<0><<<gg, 256, 0, stream>>>(xb, wqb, Qb, S_LEN, DMODEL, DMODEL);
    gemm_bf16_xwT<0><<<gg, 256, 0, stream>>>(xb, wkb, Kb, S_LEN, DMODEL, DMODEL);
    gemm_bf16_xwT<1><<<gg, 256, 0, stream>>>(xb, wvb, Vt, S_LEN, DMODEL, DMODEL);

    flash_attn_kernel<<<dim3(S_LEN / 128, NHEADS), 256, 0, stream>>>(Qb, Kb, Vt, Cb);

    gemm_bf16_xwT<2><<<gg, 256, 0, stream>>>(Cb, wob, (float*)d_out,
                                             S_LEN, DMODEL, DMODEL);
}
